// ConformerMHSARelposV1_88450556494797
// MI455X (gfx1250) — compile-verified, matrix operand order checked by reading, not run on hardware
//
#include <hip/hip_runtime.h>
#include <hip/hip_bf16.h>
#include <math.h>

// Shapes (compile-time constants from the reference)
#define BB 8
#define TT 1024
#define FF 512
#define HH 8
#define DD 64
#define BH 64      // B*H
#define MM 8192    // B*T

typedef __bf16 bf16_t;
typedef __attribute__((ext_vector_type(16))) __bf16 v16bf;
typedef __attribute__((ext_vector_type(8)))  __bf16 v8bf;
typedef __attribute__((ext_vector_type(4)))  __bf16 v4bf;
typedef __attribute__((ext_vector_type(2)))  __bf16 v2bf;
typedef __attribute__((ext_vector_type(8)))  float  v8f;
typedef __attribute__((ext_vector_type(4)))  float  v4f;

// ---------------------------------------------------------------------------
// CDNA5 async global->LDS copy (ASYNCcnt-tracked, no VGPR round trip).
// Generic pointer of a __shared__ object: low 32 bits == LDS byte offset.
// ---------------------------------------------------------------------------
__device__ __forceinline__ void async_copy_b128(void* lds_dst, const void* gsrc) {
  unsigned l = (unsigned)(uintptr_t)lds_dst;
  asm volatile("global_load_async_to_lds_b128 %0, %1, off"
               :: "v"(l), "v"(gsrc) : "memory");
}
__device__ __forceinline__ void wait_async_all() {
  asm volatile("s_wait_asynccnt 0x0" ::: "memory");
}

// ---------------------------------------------------------------------------
// WMMA helpers (CDNA5 wave32, 16x16x32 bf16 -> f32)
// A fragment (16x32, row-major source): lane holds row m = lane&15.
//   halves 0..7  = K[k0 + off .. k0+off+7]      (off = lane>=16 ? 8 : 0)
//   halves 8..15 = K[k0 + off + 16 .. +23]
// B fragment (32x16): mirrored — lane holds column n = lane&15, same K strip.
// ---------------------------------------------------------------------------
__device__ __forceinline__ v8f wmma_bf16(v16bf a, v16bf b, v8f c) {
  return __builtin_amdgcn_wmma_f32_16x16x32_bf16(
      /*neg_a=*/false, a, /*neg_b=*/false, b,
      /*c_mod=*/(short)0, c, /*reuse_a=*/false, /*reuse_b=*/false);
}

__device__ __forceinline__ v16bf frag_bf16(const bf16_t* p, bool hi) {
  p += hi ? 8 : 0;
  v8bf lo = *(const v8bf*)(p);
  v8bf hh = *(const v8bf*)(p + 16);
  v16bf r;
#pragma unroll
  for (int i = 0; i < 8; ++i) { r[i] = lo[i]; r[i + 8] = hh[i]; }
  return r;
}

__device__ __forceinline__ v16bf frag_f32(const float* p, bool hi) {
  p += hi ? 8 : 0;
  v4f a = *(const v4f*)(p);
  v4f b = *(const v4f*)(p + 4);
  v4f c = *(const v4f*)(p + 16);
  v4f d = *(const v4f*)(p + 20);
  v16bf r;
#pragma unroll
  for (int i = 0; i < 4; ++i) {
    r[i]      = (bf16_t)a[i];
    r[i + 4]  = (bf16_t)b[i];
    r[i + 8]  = (bf16_t)c[i];
    r[i + 12] = (bf16_t)d[i];
  }
  return r;
}

__device__ __forceinline__ void zero_acc(v8f* acc, int n) {
  for (int j = 0; j < n; ++j)
#pragma unroll
    for (int r = 0; r < 8; ++r) acc[j][r] = 0.0f;
}

// ---------------------------------------------------------------------------
// K0: one-time f32 -> bf16 conversion of the four 512x512 weight matrices.
// grid (256, 4), block 256, 4 elements/thread.
// ---------------------------------------------------------------------------
__global__ void wcvt_kernel(const float* __restrict__ w0, const float* __restrict__ w1,
                            const float* __restrict__ w2, const float* __restrict__ w3,
                            bf16_t* __restrict__ o0, bf16_t* __restrict__ o1,
                            bf16_t* __restrict__ o2, bf16_t* __restrict__ o3) {
  const int mat = blockIdx.y;
  const float* src = (mat == 0) ? w0 : (mat == 1) ? w1 : (mat == 2) ? w2 : w3;
  bf16_t*      dst = (mat == 0) ? o0 : (mat == 1) ? o1 : (mat == 2) ? o2 : o3;
  const int i = (blockIdx.x * 256 + threadIdx.x) * 4;
  float4 v = *(const float4*)(src + i);
  v4bf o;
  o[0] = (bf16_t)v.x; o[1] = (bf16_t)v.y; o[2] = (bf16_t)v.z; o[3] = (bf16_t)v.w;
  *(v4bf*)(dst + i) = o;
}

// ---------------------------------------------------------------------------
// K1: LayerNorm (f32 in) -> bf16 row-major (MM x FF)
// ---------------------------------------------------------------------------
__global__ void ln_kernel(const float* __restrict__ x, const float* __restrict__ g,
                          const float* __restrict__ bta, bf16_t* __restrict__ y) {
  const int row = blockIdx.x;
  const int tid = threadIdx.x;
  const float* xr = x + (size_t)row * FF;
  float2 v = *(const float2*)(xr + tid * 2);

  __shared__ float red[256];
  red[tid] = v.x + v.y;
  __syncthreads();
  for (int o = 128; o > 0; o >>= 1) { if (tid < o) red[tid] += red[tid + o]; __syncthreads(); }
  const float mu = red[0] * (1.0f / FF);
  __syncthreads();

  const float d0 = v.x - mu, d1 = v.y - mu;
  red[tid] = d0 * d0 + d1 * d1;
  __syncthreads();
  for (int o = 128; o > 0; o >>= 1) { if (tid < o) red[tid] += red[tid + o]; __syncthreads(); }
  const float rs = rsqrtf(red[0] * (1.0f / FF) + 1e-5f);

  const int c = tid * 2;
  v2bf o;
  o[0] = (bf16_t)(d0 * rs * g[c]     + bta[c]);
  o[1] = (bf16_t)(d1 * rs * g[c + 1] + bta[c + 1]);
  *(v2bf*)(y + (size_t)row * FF + c) = o;
}

// ---------------------------------------------------------------------------
// K2: QKV projection. out[i,j] = sum_k xln[i,k] * Wbf[j,k] + bias[j]
// Weight tile (64 cols x 512 K, bf16, 64KB) async-staged into LDS once per
// block; shared by all 8 waves. Q,K stored [bh][t][d]; V stored [bh][d][t].
// grid (MM/128, FF/64, 3), block 256 (8 waves, wave = 16 rows x 64 cols)
// ---------------------------------------------------------------------------
__global__ void qkv_kernel(const bf16_t* __restrict__ xln,
                           const bf16_t* __restrict__ wqb, const float* __restrict__ bq,
                           const bf16_t* __restrict__ wkb, const float* __restrict__ bk,
                           const bf16_t* __restrict__ wvb, const float* __restrict__ bv,
                           bf16_t* __restrict__ Q, bf16_t* __restrict__ K,
                           bf16_t* __restrict__ Vt) {
  const int wave = threadIdx.x >> 5;
  const int lane = threadIdx.x & 31;
  const bool hi = lane >= 16;
  const int l15 = lane & 15;
  const int m0 = blockIdx.x * 128 + wave * 16;
  const int n0 = blockIdx.y * 64;
  const int mat = blockIdx.z;
  const bf16_t* W   = (mat == 0) ? wqb : (mat == 1) ? wkb : wvb;
  const float* bias = (mat == 0) ? bq  : (mat == 1) ? bk  : bv;

  __shared__ bf16_t wtile[64 * 512];   // 64 KB
  {
    const bf16_t* wsrc = W + (size_t)n0 * FF;   // rows n0..n0+63, contiguous
#pragma unroll
    for (int it = 0; it < 16; ++it) {
      const int e = (threadIdx.x + it * 256) * 8;   // 16B chunks
      async_copy_b128(wtile + e, wsrc + e);
    }
  }
  wait_async_all();
  __syncthreads();

  v8f acc[4];
  zero_acc(acc, 4);
  const bf16_t* arow = xln + (size_t)(m0 + l15) * FF;

  for (int k0 = 0; k0 < FF; k0 += 32) {
    v16bf a = frag_bf16(arow + k0, hi);
    v16bf b[4];
#pragma unroll
    for (int j = 0; j < 4; ++j)
      b[j] = frag_bf16(wtile + (j * 16 + l15) * FF + k0, hi);
#pragma unroll
    for (int j = 0; j < 4; ++j)
      acc[j] = wmma_bf16(a, b[j], acc[j]);
  }

#pragma unroll
  for (int j = 0; j < 4; ++j) {
    const int n = n0 + j * 16 + l15;
    const int h = n >> 6, d = n & 63;
    const float bs = bias[n];
#pragma unroll
    for (int r = 0; r < 8; ++r) {
      const int i = m0 + r + (hi ? 8 : 0);
      const int b_ = i >> 10, t = i & 1023;
      const int bh = b_ * HH + h;
      const bf16_t val = (bf16_t)(acc[j][r] + bs);
      if (mat == 0)      Q[((size_t)bh * TT + t) * DD + d] = val;
      else if (mat == 1) K[((size_t)bh * TT + t) * DD + d] = val;
      else               Vt[(size_t)bh * (DD * TT) + (size_t)d * TT + t] = val;
    }
  }
}

// ---------------------------------------------------------------------------
// K3: qp term. For fixed t: scores[bh, t, s] = sum_d Q[bh,t,d] * rel[t,s,d]
// Q_t (64x64 bf16, 8KB) async-staged to LDS (shared by all waves).
// rel stays f32 (consumed once) and is converted in registers.
// grid (TT, TT/128), block 256; wave = 16 cols x 64 bh rows.
// ---------------------------------------------------------------------------
__global__ void qp_kernel(const bf16_t* __restrict__ Q, const float* __restrict__ rel,
                          float* __restrict__ scores) {
  const int wave = threadIdx.x >> 5;
  const int lane = threadIdx.x & 31;
  const bool hi = lane >= 16;
  const int l15 = lane & 15;
  const int t  = blockIdx.x;
  const int n0 = blockIdx.y * 128 + wave * 16;

  __shared__ bf16_t qtile[64 * 64];   // 8 KB
  {
#pragma unroll
    for (int it = 0; it < 2; ++it) {
      const int e = threadIdx.x + it * 256;   // 512 chunks of 16B
      const int row = e >> 3, col = (e & 7) * 8;
      async_copy_b128(qtile + row * DD + col,
                      Q + ((size_t)row * TT + t) * DD + col);
    }
  }
  wait_async_all();
  __syncthreads();

  v8f acc[4];
  zero_acc(acc, 4);
  const float* brow = rel + ((size_t)t * TT + n0 + l15) * DD;
  __builtin_prefetch(brow, 0, 1);

  for (int k0 = 0; k0 < DD; k0 += 32) {
    v16bf b = frag_f32(brow + k0, hi);
    v16bf a[4];
#pragma unroll
    for (int g = 0; g < 4; ++g)
      a[g] = frag_bf16(qtile + (g * 16 + l15) * DD + k0, hi);
#pragma unroll
    for (int g = 0; g < 4; ++g)
      acc[g] = wmma_bf16(a[g], b, acc[g]);
  }

#pragma unroll
  for (int g = 0; g < 4; ++g) {
    const int s = n0 + l15;
#pragma unroll
    for (int r = 0; r < 8; ++r) {
      const int bh = g * 16 + r + (hi ? 8 : 0);
      scores[((size_t)bh * TT + t) * TT + s] = acc[g][r];
    }
  }
}

// ---------------------------------------------------------------------------
// K4: qk term, accumulated on top of qp via the WMMA C operand.
// K tile (128 rows x 64 d, bf16, 16KB) async-staged to LDS once per block.
// scores[bh,t,s] += sum_d Q[bh,t,d]*K[bh,s,d].  grid (BH, 8, 8), block 256.
// Each wave: 16 t-rows x 128 s-cols (8 accumulators).
// ---------------------------------------------------------------------------
__global__ void qk_kernel(const bf16_t* __restrict__ Q, const bf16_t* __restrict__ K,
                          float* __restrict__ scores) {
  const int wave = threadIdx.x >> 5;
  const int lane = threadIdx.x & 31;
  const bool hi = lane >= 16;
  const int l15 = lane & 15;
  const int bh = blockIdx.x;
  const int t0 = blockIdx.y * 128 + wave * 16;
  const int s0 = blockIdx.z * 128;

  __shared__ bf16_t ktile[128 * 64];   // 16 KB
  {
    const bf16_t* ksrc = K + ((size_t)bh * TT + s0) * DD;   // 128 rows contiguous
#pragma unroll
    for (int it = 0; it < 4; ++it) {
      const int e = (threadIdx.x + it * 256) * 8;
      async_copy_b128(ktile + e, ksrc + e);
    }
  }

  v8f acc[8];   // init from qp-written scores (overlaps with async staging)
#pragma unroll
  for (int j = 0; j < 8; ++j)
#pragma unroll
    for (int r = 0; r < 8; ++r)
      acc[j][r] = scores[((size_t)bh * TT + t0 + r + (hi ? 8 : 0)) * TT + s0 + j * 16 + l15];

  wait_async_all();
  __syncthreads();

  const bf16_t* arow = Q + ((size_t)bh * TT + t0 + l15) * DD;
  for (int k0 = 0; k0 < DD; k0 += 32) {
    v16bf a = frag_bf16(arow + k0, hi);
    v16bf b[8];
#pragma unroll
    for (int j = 0; j < 8; ++j)
      b[j] = frag_bf16(ktile + (j * 16 + l15) * DD + k0, hi);
#pragma unroll
    for (int j = 0; j < 8; ++j)
      acc[j] = wmma_bf16(a, b[j], acc[j]);
  }

#pragma unroll
  for (int j = 0; j < 8; ++j)
#pragma unroll
    for (int r = 0; r < 8; ++r)
      scores[((size_t)bh * TT + t0 + r + (hi ? 8 : 0)) * TT + s0 + j * 16 + l15] = acc[j][r];
}

// ---------------------------------------------------------------------------
// K5: masked softmax over s; scale by 1/sqrt(D); write bf16 attn IN PLACE
// over the f32 score row (row stride stays 1024 f32 = 2048 bf16).
// grid (TT, BH), block 256 (4 elem/thread).
// ---------------------------------------------------------------------------
__global__ void softmax_kernel(float* __restrict__ scores,
                               const unsigned char* __restrict__ mask) {
  const int t = blockIdx.x;
  const int bh = blockIdx.y;
  const int b = bh >> 3;
  const int tid = threadIdx.x;
  float* row = scores + ((size_t)bh * TT + t) * TT;

  float4 v = *(const float4*)(row + tid * 4);
  uchar4 mk = *(const uchar4*)(mask + (size_t)b * TT + tid * 4);
  const float MIN_F = -3.402823466e38f;
  const float sc = 0.125f;  // 1/sqrt(64)
  float x0 = mk.x ? v.x * sc : MIN_F;
  float x1 = mk.y ? v.y * sc : MIN_F;
  float x2 = mk.z ? v.z * sc : MIN_F;
  float x3 = mk.w ? v.w * sc : MIN_F;

  __shared__ float red[256];
  red[tid] = fmaxf(fmaxf(x0, x1), fmaxf(x2, x3));
  __syncthreads();
  for (int o = 128; o > 0; o >>= 1) { if (tid < o) red[tid] = fmaxf(red[tid], red[tid + o]); __syncthreads(); }
  const float mx = red[0];
  __syncthreads();

  const float e0 = expf(x0 - mx), e1 = expf(x1 - mx);
  const float e2 = expf(x2 - mx), e3 = expf(x3 - mx);
  red[tid] = e0 + e1 + e2 + e3;
  __syncthreads();
  for (int o = 128; o > 0; o >>= 1) { if (tid < o) red[tid] += red[tid + o]; __syncthreads(); }
  const float inv = 1.0f / red[0];

  v4bf o;
  o[0] = (bf16_t)(mk.x ? e0 * inv : 0.0f);
  o[1] = (bf16_t)(mk.y ? e1 * inv : 0.0f);
  o[2] = (bf16_t)(mk.z ? e2 * inv : 0.0f);
  o[3] = (bf16_t)(mk.w ? e3 * inv : 0.0f);
  *(v4bf*)((bf16_t*)row + tid * 4) = o;   // safe: all reads barriered above
}

// ---------------------------------------------------------------------------
// K6: ctx = attn @ V.  attn = bf16 rows (stride 2048) inside the score
// buffer.  V chunks (64 d-rows x 128 s, 16KB) are double-buffered in LDS via
// async copies: chunk c+1 streams in while WMMAs consume chunk c.
// grid (BH, 8), block 256; wave = 16 t-rows x 64 d-cols.
// ---------------------------------------------------------------------------
__global__ void av_kernel(const float* __restrict__ scores, const bf16_t* __restrict__ Vt,
                          bf16_t* __restrict__ ctx) {
  const int wave = threadIdx.x >> 5;
  const int lane = threadIdx.x & 31;
  const bool hi = lane >= 16;
  const int l15 = lane & 15;
  const int bh = blockIdx.x;
  const int t0 = blockIdx.y * 128 + wave * 16;

  __shared__ bf16_t vbuf[2][64 * 128];   // 2 x 16 KB
  const bf16_t* vbase = Vt + (size_t)bh * (DD * TT);

  auto stage = [&](int buf, int c) {
#pragma unroll
    for (int it = 0; it < 4; ++it) {
      const int e = threadIdx.x + it * 256;      // 1024 chunks of 16B
      const int row = e >> 4, col = (e & 15) * 8;
      async_copy_b128(&vbuf[buf][row * 128 + col],
                      vbase + (size_t)row * TT + c * 128 + col);
    }
  };

  stage(0, 0);
  wait_async_all();
  __syncthreads();

  v8f acc[4];
  zero_acc(acc, 4);
  const bf16_t* attn = (const bf16_t*)scores;
  const bf16_t* arow = attn + ((size_t)bh * TT + t0 + l15) * (2 * TT);

  for (int c = 0; c < 8; ++c) {
    if (c + 1 < 8) stage((c + 1) & 1, c + 1);
    const bf16_t* vb = vbuf[c & 1];
#pragma unroll
    for (int kk = 0; kk < 128; kk += 32) {
      const int k0 = c * 128 + kk;
      __builtin_prefetch(arow + k0 + 128, 0, 1);
      v16bf a = frag_bf16(arow + k0, hi);
      v16bf b[4];
#pragma unroll
      for (int j = 0; j < 4; ++j)
        b[j] = frag_bf16(vb + (j * 16 + l15) * 128 + kk, hi);
#pragma unroll
      for (int j = 0; j < 4; ++j)
        acc[j] = wmma_bf16(a, b[j], acc[j]);
    }
    wait_async_all();
    __syncthreads();
  }

  const int b = bh >> 3, h = bh & 7;
#pragma unroll
  for (int j = 0; j < 4; ++j) {
    const int d = j * 16 + l15;
#pragma unroll
    for (int r = 0; r < 8; ++r) {
      const int t = t0 + r + (hi ? 8 : 0);
      ctx[((size_t)(b * TT + t)) * FF + h * DD + d] = (bf16_t)acc[j][r];
    }
  }
}

// ---------------------------------------------------------------------------
// K7: final projection: out = ctx @ wo^T + bo (f32 out).  wo tile (bf16,
// 64KB) async-staged to LDS. grid (MM/128, FF/64), block 256.
// ---------------------------------------------------------------------------
__global__ void out_kernel(const bf16_t* __restrict__ ctx, const bf16_t* __restrict__ wob,
                           const float* __restrict__ bo, float* __restrict__ out) {
  const int wave = threadIdx.x >> 5;
  const int lane = threadIdx.x & 31;
  const bool hi = lane >= 16;
  const int l15 = lane & 15;
  const int m0 = blockIdx.x * 128 + wave * 16;
  const int n0 = blockIdx.y * 64;

  __shared__ bf16_t wtile[64 * 512];   // 64 KB
  {
    const bf16_t* wsrc = wob + (size_t)n0 * FF;
#pragma unroll
    for (int it = 0; it < 16; ++it) {
      const int e = (threadIdx.x + it * 256) * 8;
      async_copy_b128(wtile + e, wsrc + e);
    }
  }
  wait_async_all();
  __syncthreads();

  v8f acc[4];
  zero_acc(acc, 4);
  const bf16_t* arow = ctx + (size_t)(m0 + l15) * FF;

  for (int k0 = 0; k0 < FF; k0 += 32) {
    v16bf a = frag_bf16(arow + k0, hi);
    v16bf b[4];
#pragma unroll
    for (int j = 0; j < 4; ++j)
      b[j] = frag_bf16(wtile + (j * 16 + l15) * FF + k0, hi);
#pragma unroll
    for (int j = 0; j < 4; ++j)
      acc[j] = wmma_bf16(a, b[j], acc[j]);
  }

#pragma unroll
  for (int j = 0; j < 4; ++j) {
    const int n = n0 + j * 16 + l15;
    const float bs = bo[n];
#pragma unroll
    for (int r = 0; r < 8; ++r) {
      const int i = m0 + r + (hi ? 8 : 0);
      out[(size_t)i * FF + n] = acc[j][r] + bs;
    }
  }
}

// ---------------------------------------------------------------------------
extern "C" void kernel_launch(void* const* d_in, const int* in_sizes, int n_in,
                              void* d_out, int out_size, void* d_ws, size_t ws_size,
                              hipStream_t stream) {
  (void)in_sizes; (void)n_in; (void)out_size; (void)ws_size;

  const float* x    = (const float*)d_in[0];   // (B,T,F)
  const float* rel  = (const float*)d_in[1];   // (T,T,D)
  const unsigned char* msk = (const unsigned char*)d_in[2]; // (B,T) bool
  const float* lns  = (const float*)d_in[3];
  const float* lnb  = (const float*)d_in[4];
  const float* wq   = (const float*)d_in[5];
  const float* bq   = (const float*)d_in[6];
  const float* wk   = (const float*)d_in[7];
  const float* bk   = (const float*)d_in[8];
  const float* wv   = (const float*)d_in[9];
  const float* bv   = (const float*)d_in[10];
  const float* wo   = (const float*)d_in[11];
  const float* bo   = (const float*)d_in[12];
  float* out = (float*)d_out;

  char* ws = (char*)d_ws;
  bf16_t* xln = (bf16_t*)(ws);                           //  8 MB
  bf16_t* Q   = (bf16_t*)(ws + ((size_t)8   << 20));     //  8 MB
  bf16_t* K   = (bf16_t*)(ws + ((size_t)16  << 20));     //  8 MB
  bf16_t* Vt  = (bf16_t*)(ws + ((size_t)24  << 20));     //  8 MB
  bf16_t* ctx = (bf16_t*)(ws + ((size_t)32  << 20));     //  8 MB
  float*  sc  = (float*)(ws + ((size_t)40  << 20));      // 256 MB
  bf16_t* wqb = (bf16_t*)(ws + ((size_t)296 << 20));     // 512 KB each
  bf16_t* wkb = wqb + (size_t)FF * FF;
  bf16_t* wvb = wkb + (size_t)FF * FF;
  bf16_t* wob = wvb + (size_t)FF * FF;

  wcvt_kernel<<<dim3(256, 4), 256, 0, stream>>>(wq, wk, wv, wo, wqb, wkb, wvb, wob);
  ln_kernel<<<MM, 256, 0, stream>>>(x, lns, lnb, xln);
  qkv_kernel<<<dim3(MM / 128, FF / 64, 3), 256, 0, stream>>>(
      xln, wqb, bq, wkb, bk, wvb, bv, Q, K, Vt);
  qp_kernel<<<dim3(TT, TT / 128), 256, 0, stream>>>(Q, rel, sc);
  qk_kernel<<<dim3(BH, TT / 128, TT / 128), 256, 0, stream>>>(Q, K, sc);
  softmax_kernel<<<dim3(TT, BH), 256, 0, stream>>>(sc, msk);
  av_kernel<<<dim3(BH, TT / 128), 256, 0, stream>>>(sc, Vt, ctx);
  out_kernel<<<dim3(MM / 128, FF / 64), 256, 0, stream>>>(ctx, wob, bo, out);
}
